// EncoderLayerRelPos2D_68478958567576
// MI455X (gfx1250) — matrix operand order repaired and checked by reading, not compiled, not run on hardware
//
#include <hip/hip_runtime.h>
#include <hip/hip_bf16.h>
#include <math.h>

// ---------------- problem constants (match reference) ----------------
#define BB    2
#define TT    2048
#define DD    768
#define HH    12
#define HDIM  64
#define FFD   3072
#define NTROW (BB * TT)          // 4096 GEMM rows
#define SPAN  63                 // 2G-1
#define SPANC 129                // 4G+1

typedef __attribute__((ext_vector_type(16))) _Float16 v16h;
typedef __attribute__((ext_vector_type(8)))  _Float16 h8;
typedef __attribute__((ext_vector_type(2)))  _Float16 h2;
typedef __attribute__((ext_vector_type(8)))  float    v8f;
typedef unsigned int u32x4 __attribute__((ext_vector_type(4)));
typedef int          i32x4 __attribute__((ext_vector_type(4)));
typedef int          i32x8 __attribute__((ext_vector_type(8)));

#if defined(__AMDGCN__) && __has_builtin(__builtin_amdgcn_tensor_load_to_lds)
#define HAVE_TDM 1
#else
#define HAVE_TDM 0
#endif

// ---------------- TDM 2D tile load: global -> padded LDS rows --------
// Loads tile_h rows x 64 halfs into LDS with an 8-half pad per row
// (72-half row stride): pad_interval=4 (32 DWORDs), pad_amount=3 (4 DWORDs).
#if HAVE_TDM
__device__ inline void tdm_load_tile64(void* lds_dst, const _Float16* gsrc,
                                       unsigned rows, unsigned tile_h,
                                       unsigned long long row_elems,
                                       unsigned long long stride_elems) {
  const unsigned long long ga = (unsigned long long)(size_t)gsrc;
  const unsigned long long s0 = stride_elems;          // dim0 stride (elems)
  u32x4 g0;
  g0[0] = 1u;                                          // count=1 (valid), user
  g0[1] = (unsigned)(size_t)lds_dst;                   // lds_addr (bytes)
  g0[2] = (unsigned)(ga & 0xffffffffu);                // global_addr[31:0]
  g0[3] = (unsigned)((ga >> 32) & 0x1ffffffu) | (2u << 30);   // [56:32], type=2
  i32x8 g1;
  g1[0] = (1 << 16) | (1 << 20) | (4 << 22) | (3 << 25);      // 2B, pad cfg
  g1[1] = (int)((row_elems & 0xffffu) << 16);                  // tensor_dim0 lo
  g1[2] = (int)((row_elems >> 16) & 0xffffu) |                 // tensor_dim0 hi
          (int)((rows & 0xffffu) << 16);                       // tensor_dim1 lo
  g1[3] = (int)((rows >> 16) & 0xffffu) | (int)(64u << 16);    // tile_dim0=64
  g1[4] = (int)tile_h;                                         // tile_dim1
  g1[5] = (int)(s0 & 0xffffffffu);                             // dim0 stride lo
  g1[6] = (int)((s0 >> 32) & 0xffffu);                         // dim0 stride hi
  g1[7] = 0;
  i32x4 z4; z4[0] = 0; z4[1] = 0; z4[2] = 0; z4[3] = 0;
#if __clang_major__ >= 23
  i32x8 z8; for (int i = 0; i < 8; ++i) z8[i] = 0;
  __builtin_amdgcn_tensor_load_to_lds(g0, g1, z4, z4, z8, 0);
#else
  __builtin_amdgcn_tensor_load_to_lds(g0, g1, z4, z4, 0);
#endif
}
#endif

// ---------------- WMMA helper (CDNA5 v_wmma_f32_16x16x32_f16) --------
__device__ inline v8f wmma_16x16x32(v16h a, v16h b, v8f c) {
  return __builtin_amdgcn_wmma_f32_16x16x32_f16(false, a, false, b,
                                                (short)0, c, false, false);
}

// A-matrix 16x32 fragment (ISA 7.12.2): lane holds row M=lane&15,
// K chunks [(lane>>4)*8,+8) and [16+(lane>>4)*8,+8). Two ds_load_b128.
__device__ inline v16h load_a_frag(const _Float16* p, int lda) {
  const int lane = threadIdx.x & 31;
  const _Float16* r = p + (lane & 15) * lda + (lane >> 4) * 8;
  h8 x0 = *(const h8*)r;
  h8 x1 = *(const h8*)(r + 16);
  return __builtin_shufflevector(x0, x1, 0, 1, 2, 3, 4, 5, 6, 7,
                                 8, 9, 10, 11, 12, 13, 14, 15);
}

// B-matrix 32x16 fragment from a K-transposed LDS tile [n][k]:
// lane holds col N=lane&15, K chunks [(lane>>4)*8,+8) and [16+(lane>>4)*8,+8).
__device__ inline v16h load_b_frag_t(const _Float16* p, int ld) {
  const int lane = threadIdx.x & 31;
  const _Float16* r = p + (lane & 15) * ld + (lane >> 4) * 8;
  h8 x0 = *(const h8*)r;
  h8 x1 = *(const h8*)(r + 16);
  return __builtin_shufflevector(x0, x1, 0, 1, 2, 3, 4, 5, 6, 7,
                                 8, 9, 10, 11, 12, 13, 14, 15);
}

__device__ inline int iclip(int v, int lo, int hi) {
  return v < lo ? lo : (v > hi ? hi : v);
}

// ---------------- elementwise f32 -> f16 -----------------------------
__global__ void cvt_f32_f16(const float* __restrict__ in,
                            _Float16* __restrict__ out, long n) {
  long i = (long)blockIdx.x * blockDim.x + threadIdx.x;
  const long stride = (long)gridDim.x * blockDim.x;
  for (; i < n; i += stride) out[i] = (_Float16)in[i];
}

// ---------------- LayerNorm (row of 768) -> f16 ----------------------
__global__ __launch_bounds__(256)
void ln_to_f16(const float* __restrict__ x, const float* __restrict__ w,
               const float* __restrict__ b, _Float16* __restrict__ out) {
  __shared__ float red[2][8];
  const int r = blockIdx.x;
  const float* xr = x + (size_t)r * DD;
  float v0[3], s = 0.f, sq = 0.f;
#pragma unroll
  for (int i = 0; i < 3; ++i) {
    float v = xr[threadIdx.x + i * 256];
    v0[i] = v; s += v; sq += v * v;
  }
#pragma unroll
  for (int m = 16; m >= 1; m >>= 1) {
    s  += __shfl_xor(s,  m, 32);
    sq += __shfl_xor(sq, m, 32);
  }
  const int wv = threadIdx.x >> 5, lane = threadIdx.x & 31;
  if (lane == 0) { red[0][wv] = s; red[1][wv] = sq; }
  __syncthreads();
  if (threadIdx.x == 0) {
    float ts = 0.f, tq = 0.f;
    for (int i = 0; i < 8; ++i) { ts += red[0][i]; tq += red[1][i]; }
    float mean = ts / DD;
    float var  = tq / DD - mean * mean;
    red[0][0] = mean;
    red[1][0] = rsqrtf(var + 1e-5f);
  }
  __syncthreads();
  const float mean = red[0][0], inv = red[1][0];
#pragma unroll
  for (int i = 0; i < 3; ++i) {
    const int c = threadIdx.x + i * 256;
    out[(size_t)r * DD + c] = (_Float16)((v0[i] - mean) * inv * w[c] + b[c]);
  }
}

// ---------------- WMMA GEMM: C(MxN) = A(MxK) @ B(KxN) ----------------
// block 256 = 8 waves; tile 128x128, K-step 64; 16 WMMAs/wave/step.
// A-tile staged by the Tensor Data Mover (padded rows, TENSORcnt);
// B-tile staged K-transposed ([n][k]) via packed b32 stores.
template <int FB, int FG, int FR, int FH>
__global__ __launch_bounds__(256)
void gemm_wmma(const _Float16* __restrict__ A, const _Float16* __restrict__ Bw,
               const float* __restrict__ bias, const float* __restrict__ resid,
               float* __restrict__ Cf, _Float16* __restrict__ Ch,
               int M, int N, int K) {
  __shared__ _Float16 As[128 * 72];   // [m][k], k=0..63, pad 8
  __shared__ _Float16 Bs[128 * 72];   // [n][k], k=0..63, pad 8
  const int tid  = threadIdx.x;
  const int lane = tid & 31;
  const int w    = tid >> 5;
  const int wm   = w >> 1, wn = w & 1;
  const int m0   = blockIdx.y * 128;
  const int n0   = blockIdx.x * 128;

  v8f acc[2][4];
#pragma unroll
  for (int i = 0; i < 2; ++i)
#pragma unroll
    for (int j = 0; j < 4; ++j) acc[i][j] = v8f{};

  const int ar  = tid >> 1, aseg = (tid & 1) * 32;       // A fallback staging
  const int bk  = (tid >> 3) * 2, bseg = (tid & 7) * 16; // B staging

  for (int k0 = 0; k0 < K; k0 += 64) {
#if HAVE_TDM
    if (w == 0)   // one wave drives the DMA: 128 rows x 64 halfs, 72-half pitch
      tdm_load_tile64(As, A + (size_t)m0 * K + k0, 128, 128,
                      (unsigned long long)K, (unsigned long long)K);
#else
    {
      const h8* src = (const h8*)(A + (size_t)(m0 + ar) * K + k0 + aseg);
      h8* dst = (h8*)(&As[ar * 72 + aseg]);
      dst[0] = src[0]; dst[1] = src[1]; dst[2] = src[2]; dst[3] = src[3];
    }
#endif
    { // B: transpose 2 k-rows x 16 cols into [n][k] with b32 stores
      const _Float16* s0 = Bw + (size_t)(k0 + bk) * N + n0 + bseg;
      const _Float16* s1 = s0 + N;
      h8 r0a = ((const h8*)s0)[0], r0b = ((const h8*)s0)[1];
      h8 r1a = ((const h8*)s1)[0], r1b = ((const h8*)s1)[1];
#pragma unroll
      for (int e = 0; e < 8; ++e) {
        *(h2*)&Bs[(bseg + e)     * 72 + bk] = h2{r0a[e], r1a[e]};
        *(h2*)&Bs[(bseg + 8 + e) * 72 + bk] = h2{r0b[e], r1b[e]};
      }
    }
    if (k0 + 64 < K) { // hint the next B K-slab toward L2
      __builtin_prefetch(Bw + (size_t)(k0 + 64 + bk) * N + n0 + bseg, 0, 0);
    }
#if HAVE_TDM
    if (w == 0) __builtin_amdgcn_s_wait_tensorcnt(0);
#endif
    __syncthreads();
    v16h af[2][2];
#pragma unroll
    for (int kc = 0; kc < 2; ++kc) {
      af[0][kc] = load_a_frag(&As[(wm * 32 + 0)  * 72 + kc * 32], 72);
      af[1][kc] = load_a_frag(&As[(wm * 32 + 16) * 72 + kc * 32], 72);
    }
#pragma unroll
    for (int ns = 0; ns < 4; ++ns) {
#pragma unroll
      for (int kc = 0; kc < 2; ++kc) {
        v16h bf = load_b_frag_t(&Bs[(wn * 64 + ns * 16) * 72 + kc * 32], 72);
        acc[0][ns] = wmma_16x16x32(af[0][kc], bf, acc[0][ns]);
        acc[1][ns] = wmma_16x16x32(af[1][kc], bf, acc[1][ns]);
      }
    }
    __syncthreads();
  }

#pragma unroll
  for (int ms = 0; ms < 2; ++ms)
#pragma unroll
    for (int ns = 0; ns < 4; ++ns) {
      const int mb = m0 + wm * 32 + ms * 16 + 8 * (lane >> 4);
      const int n  = n0 + wn * 64 + ns * 16 + (lane & 15);
#pragma unroll
      for (int i = 0; i < 8; ++i) {
        const int m = mb + i;
        float v = acc[ms][ns][i];
        if (FB) v += bias[n];
        if (FG) v = 0.5f * v * (1.0f + erff(v * 0.70710678118654752f));
        if (FR) v += resid[(size_t)m * N + n];
        if (FH) Ch[(size_t)m * N + n] = (_Float16)v;
        else    Cf[(size_t)m * N + n] = v;
      }
    }
}

// ---------------- flash attention with inline rel/demo bias ----------
// grid: (T/64, H, B); block 128 = 4 waves; wave wv owns q rows [16wv,+16).
// Q and K tiles staged by TDM (strided 2D tiles, natural layouts);
// V tile staged transposed [hd][key] manually for the P.V B-operand.
__global__ __launch_bounds__(128)
void attn_flash(const _Float16* __restrict__ qkv,   // (B*T, 3D) f16
                const float* __restrict__ rel_emb,  // (63*63, 12)
                const float* __restrict__ demo_emb, // (129*63, 12)
                const int* __restrict__ row,  const int* __restrict__ col,
                const int* __restrict__ drow, const int* __restrict__ dcol,
                const int* __restrict__ did,
                const unsigned char* __restrict__ sep,
                _Float16* __restrict__ out)         // (B*T, D) f16
{
  __shared__ _Float16 Qs[64 * 72];        // [q][hd]
  __shared__ _Float16 Ks[64 * 72];        // [key][hd] (natural layout)
  __shared__ _Float16 Vs[64 * 72];        // [hd][key] (transposed)
  __shared__ _Float16 Ps[4][16 * 72];     // per-wave P tile [qrow][key]
  __shared__ int q_row[64], q_col[64], q_dr[64], q_dc[64], q_id[64], q_sp[64];
  __shared__ int k_rowS[64], k_colS[64], k_drS[64], k_dcS[64], k_idS[64], k_spS[64];

  const int tid = threadIdx.x, lane = tid & 31, wv = tid >> 5;
  const int half = lane >> 4, nl = lane & 15;
  const int qt = blockIdx.x, h = blockIdx.y, bz = blockIdx.z;
  const size_t rs3 = 3 * DD;
  const _Float16* qbase = qkv + ((size_t)bz * TT + qt * 64) * rs3 + h * HDIM;

#if HAVE_TDM
  if (wv == 0)
    tdm_load_tile64(Qs, qbase, 64, 64, rs3, rs3);
#else
  { // stage Q tile (64x64)
    const int r = tid >> 1, seg = (tid & 1) * 32;
    const h8* src = (const h8*)(qbase + (size_t)r * rs3 + seg);
    h8* dst = (h8*)(&Qs[r * 72 + seg]);
    dst[0] = src[0]; dst[1] = src[1]; dst[2] = src[2]; dst[3] = src[3];
  }
#endif
  if (tid < 64) {
    const int g = qt * 64 + tid;
    q_row[tid] = row[g];  q_col[tid] = col[g];
    q_dr[tid]  = drow[g]; q_dc[tid]  = dcol[g];
    q_id[tid]  = did[g];  q_sp[tid]  = sep[g];
  }
#if HAVE_TDM
  if (wv == 0) __builtin_amdgcn_s_wait_tensorcnt(0);
#endif
  __syncthreads();

  // hoist q-side metadata (loop-invariant) into registers: row m=i+8*half
  int qrw[8], qcl[8], qdrr[8], qdcc[8], qidd[8], qspp[8];
#pragma unroll
  for (int i = 0; i < 8; ++i) {
    const int idx = wv * 16 + i + 8 * half;
    qrw[i] = q_row[idx]; qcl[i] = q_col[idx];
    qdrr[i] = q_dr[idx]; qdcc[i] = q_dc[idx];
    qidd[i] = q_id[idx]; qspp[i] = q_sp[idx];
  }

  float mrun[8], lrun[8];
#pragma unroll
  for (int i = 0; i < 8; ++i) { mrun[i] = -1e30f; lrun[i] = 0.f; }
  v8f o[4];
#pragma unroll
  for (int i = 0; i < 4; ++i) o[i] = v8f{};
  const float scale = 0.125f;   // 1/sqrt(64)

  for (int j = 0; j < TT / 64; ++j) {
    __syncthreads();            // prior tile's WMMA reads done
    const _Float16* kbase = qkv + ((size_t)bz * TT + j * 64) * rs3 + DD + h * HDIM;
#if HAVE_TDM
    if (wv == 0)                // K tile: strided 2D copy, natural layout
      tdm_load_tile64(Ks, kbase, 64, 64, rs3, rs3);
#else
    { // stage K row-major (straight copy)
      const int r = tid >> 1, seg = (tid & 1) * 32;
      const h8* src = (const h8*)(kbase + (size_t)r * rs3 + seg);
      h8* dst = (h8*)(&Ks[r * 72 + seg]);
      dst[0] = src[0]; dst[1] = src[1]; dst[2] = src[2]; dst[3] = src[3];
    }
#endif
    { // stage V transposed: 2 key-rows x 16 hd-cols -> [hd][key] b32 stores
      const int kp = (tid >> 2) * 2, seg = (tid & 3) * 16;
      const _Float16* v0 = qkv + ((size_t)bz * TT + j * 64 + kp) * rs3 + 2 * DD + h * HDIM + seg;
      const _Float16* v1 = v0 + rs3;
      h8 r0a = ((const h8*)v0)[0], r0b = ((const h8*)v0)[1];
      h8 r1a = ((const h8*)v1)[0], r1b = ((const h8*)v1)[1];
#pragma unroll
      for (int e = 0; e < 8; ++e) {
        *(h2*)&Vs[(seg + e)     * 72 + kp] = h2{r0a[e], r1a[e]};
        *(h2*)&Vs[(seg + 8 + e) * 72 + kp] = h2{r0b[e], r1b[e]};
      }
    }
    if (tid < 64) {
      const int g = j * 64 + tid;
      k_rowS[tid] = row[g];  k_colS[tid] = col[g];
      k_drS[tid]  = drow[g]; k_dcS[tid]  = dcol[g];
      k_idS[tid]  = did[g];  k_spS[tid]  = sep[g];
    }
#if HAVE_TDM
    if (wv == 0) __builtin_amdgcn_s_wait_tensorcnt(0);
#endif
    __syncthreads();

    // S = Q @ K^T (16x64 per wave); K tile row-major is the B layout
    v8f s[4];
#pragma unroll
    for (int nt = 0; nt < 4; ++nt) {
      s[nt] = v8f{};
#pragma unroll
      for (int kc = 0; kc < 64; kc += 32) {
        v16h a  = load_a_frag(&Qs[(wv * 16) * 72 + kc], 72);
        v16h bm = load_b_frag_t(&Ks[(nt * 16) * 72 + kc], 72);
        s[nt] = wmma_16x16x32(a, bm, s[nt]);
      }
    }

    // hoist k-side metadata per nt (this lane's column nl only)
    int krw[4], kcl[4], kdr[4], kdc[4], kid[4], ksp[4];
#pragma unroll
    for (int nt = 0; nt < 4; ++nt) {
      const int kn = nt * 16 + nl;
      krw[nt] = k_rowS[kn]; kcl[nt] = k_colS[kn];
      kdr[nt] = k_drS[kn];  kdc[nt] = k_dcS[kn];
      kid[nt] = k_idS[kn];  ksp[nt] = k_spS[kn];
    }

    // inline bias + online softmax (row m = i + 8*half, col = nt*16 + nl)
    float mnew[8];
#pragma unroll
    for (int i = 0; i < 8; ++i) {
      float rowmax = -1e30f;
#pragma unroll
      for (int nt = 0; nt < 4; ++nt) {
        float bsum = 0.f;
        if (!qspp[i] && !ksp[nt]) {
          const int dr = iclip(qrw[i] - krw[nt], -31, 31) + 31;
          const int dc = iclip(qcl[i] - kcl[nt], -31, 31) + 31;
          bsum = rel_emb[(dr * SPAN + dc) * HH + h];
          if (qidd[i] == kid[nt] && qidd[i] >= 0) {
            const int dr2 = iclip(qdrr[i] - kdr[nt], -31, 31) + 31;
            const int dc2 = iclip(qdcc[i] - kdc[nt], -64, 64) + 64;
            bsum += demo_emb[(dr2 * SPANC + dc2) * HH + h];
          }
        }
        const float lg = fmaf(s[nt][i], scale, bsum);
        s[nt][i] = lg;
        rowmax = fmaxf(rowmax, lg);
      }
#pragma unroll
      for (int msk = 1; msk < 16; msk <<= 1)
        rowmax = fmaxf(rowmax, __shfl_xor(rowmax, msk, 32));
      mnew[i] = fmaxf(mrun[i], rowmax);
    }
#pragma unroll
    for (int i = 0; i < 8; ++i) {
      const float cf = __expf(mrun[i] - mnew[i]);
      float rsum = 0.f;
#pragma unroll
      for (int nt = 0; nt < 4; ++nt) {
        const float p = __expf(s[nt][i] - mnew[i]);
        s[nt][i] = p;
        rsum += p;
      }
#pragma unroll
      for (int msk = 1; msk < 16; msk <<= 1)
        rsum += __shfl_xor(rsum, msk, 32);
      lrun[i] = lrun[i] * cf + rsum;
      mrun[i] = mnew[i];
#pragma unroll
      for (int nt = 0; nt < 4; ++nt) o[nt][i] *= cf;
      const int ml = i + 8 * half;
#pragma unroll
      for (int nt = 0; nt < 4; ++nt)
        Ps[wv][ml * 72 + nt * 16 + nl] = (_Float16)s[nt][i];
    }
    // same-wave LDS RAW: P just written, read below as A-fragments
    asm volatile("s_wait_dscnt 0" ::: "memory");

    // O += P @ V  (V transposed tile is the B layout)
#pragma unroll
    for (int nt = 0; nt < 4; ++nt) {
#pragma unroll
      for (int kc = 0; kc < 64; kc += 32) {
        v16h a  = load_a_frag(&Ps[wv][kc], 72);
        v16h bm = load_b_frag_t(&Vs[(nt * 16) * 72 + kc], 72);
        o[nt] = wmma_16x16x32(a, bm, o[nt]);
      }
    }
  }

  // normalize and write (B*T, D) f16 head slice
#pragma unroll
  for (int nt = 0; nt < 4; ++nt)
#pragma unroll
    for (int i = 0; i < 8; ++i) {
      const int t = qt * 64 + wv * 16 + i + 8 * half;
      const int n = h * HDIM + nt * 16 + nl;
      out[((size_t)bz * TT + t) * DD + n] = (_Float16)(o[nt][i] / lrun[i]);
    }
}

// ---------------- launcher ------------------------------------------
extern "C" void kernel_launch(void* const* d_in, const int* in_sizes, int n_in,
                              void* d_out, int out_size, void* d_ws, size_t ws_size,
                              hipStream_t stream) {
  const float* x       = (const float*)d_in[0];
  const int*   row     = (const int*)d_in[1];
  const int*   col     = (const int*)d_in[2];
  const int*   drow    = (const int*)d_in[3];
  const int*   dcol    = (const int*)d_in[4];
  const int*   did     = (const int*)d_in[5];
  const unsigned char* sep = (const unsigned char*)d_in[6];
  const float* ln1_w   = (const float*)d_in[7];
  const float* ln1_b   = (const float*)d_in[8];
  const float* ln2_w   = (const float*)d_in[9];
  const float* ln2_b   = (const float*)d_in[10];
  const float* qkv_w   = (const float*)d_in[11];
  const float* qkv_b   = (const float*)d_in[12];
  const float* proj_w  = (const float*)d_in[13];
  const float* proj_b  = (const float*)d_in[14];
  const float* ff1_w   = (const float*)d_in[15];
  const float* ff1_b   = (const float*)d_in[16];
  const float* ff2_w   = (const float*)d_in[17];
  const float* ff2_b   = (const float*)d_in[18];
  const float* rel_emb = (const float*)d_in[19];
  const float* demo_emb= (const float*)d_in[20];
  float* outp = (float*)d_out;

  char* ws = (char*)d_ws;
  size_t off = 0;
  auto carve = [&](size_t bytes) -> char* {
    char* p = ws + off;
    off += (bytes + 255) & ~(size_t)255;
    return p;
  };
  _Float16* hbuf    = (_Float16*)carve((size_t)NTROW * DD * 2);
  _Float16* qkvbuf  = (_Float16*)carve((size_t)NTROW * 3 * DD * 2);
  _Float16* attnbuf = (_Float16*)carve((size_t)NTROW * DD * 2);
  _Float16* midbuf  = (_Float16*)carve((size_t)NTROW * FFD * 2);
  _Float16* wqkv    = (_Float16*)carve((size_t)DD * 3 * DD * 2);
  _Float16* wproj   = (_Float16*)carve((size_t)DD * DD * 2);
  _Float16* wff1    = (_Float16*)carve((size_t)DD * FFD * 2);
  _Float16* wff2    = (_Float16*)carve((size_t)FFD * DD * 2);

  cvt_f32_f16<<<1024, 256, 0, stream>>>(qkv_w,  wqkv,  (long)DD * 3 * DD);
  cvt_f32_f16<<<512,  256, 0, stream>>>(proj_w, wproj, (long)DD * DD);
  cvt_f32_f16<<<1024, 256, 0, stream>>>(ff1_w,  wff1,  (long)DD * FFD);
  cvt_f32_f16<<<1024, 256, 0, stream>>>(ff2_w,  wff2,  (long)FFD * DD);

  // h1 = LN1(x) -> f16
  ln_to_f16<<<NTROW, 256, 0, stream>>>(x, ln1_w, ln1_b, hbuf);

  // qkv = h1 @ qkv_w + qkv_b -> f16
  gemm_wmma<1, 0, 0, 1><<<dim3(3 * DD / 128, NTROW / 128), 256, 0, stream>>>(
      hbuf, wqkv, qkv_b, nullptr, nullptr, qkvbuf, NTROW, 3 * DD, DD);

  // flash attention with inline rel/demo bias -> f16
  attn_flash<<<dim3(TT / 64, HH, BB), 128, 0, stream>>>(
      qkvbuf, rel_emb, demo_emb, row, col, drow, dcol, did, sep, attnbuf);

  // x1 = x + attn @ proj_w + proj_b -> f32 (d_out)
  gemm_wmma<1, 0, 1, 0><<<dim3(DD / 128, NTROW / 128), 256, 0, stream>>>(
      attnbuf, wproj, proj_b, x, outp, nullptr, NTROW, DD, DD);

  // h2 = LN2(x1) -> f16
  ln_to_f16<<<NTROW, 256, 0, stream>>>(outp, ln2_w, ln2_b, hbuf);

  // mid = gelu(h2 @ ff1_w + ff1_b) -> f16
  gemm_wmma<1, 1, 0, 1><<<dim3(FFD / 128, NTROW / 128), 256, 0, stream>>>(
      hbuf, wff1, ff1_b, nullptr, nullptr, midbuf, NTROW, FFD, DD);

  // out = x1 + mid @ ff2_w + ff2_b -> f32 (in-place on d_out)
  gemm_wmma<1, 0, 1, 0><<<dim3(DD / 128, NTROW / 128), 256, 0, stream>>>(
      midbuf, wff2, ff2_b, outp, outp, nullptr, NTROW, DD, FFD);

  (void)in_sizes; (void)n_in; (void)out_size; (void)ws_size;
}
